// DistillLossCQColBERT_74466142978317
// MI455X (gfx1250) — compile-verified, matrix operand order checked by reading, not run on hardware
//
#include <hip/hip_runtime.h>

typedef __attribute__((ext_vector_type(2)))  float  v2f;
typedef __attribute__((ext_vector_type(8)))  float  v8f;
typedef __attribute__((ext_vector_type(16))) __bf16 v16bf;

// Exact pointee type of __builtin_amdgcn_global_load_async_to_lds_b128 params:
// (int4 addrspace(1)* src, int4 addrspace(3)* lds, imm offset, imm cpol)
typedef int v4i_ __attribute__((__vector_size__(16)));
typedef __attribute__((address_space(1))) v4i_ gv4i;   // global int4
typedef __attribute__((address_space(3))) v4i_ lv4i;   // LDS int4

#if __has_builtin(__builtin_amdgcn_global_load_async_to_lds_b128) && \
    __has_builtin(__builtin_amdgcn_s_wait_asynccnt)
#define USE_ASYNC 1
#else
#define USE_ASYNC 0
#endif

static constexpr int kB  = 128;   // batch
static constexpr int kN  = 8;     // num docs
static constexpr int kLq = 32;    // query tokens
static constexpr int kLd = 256;   // doc tokens
static constexpr int kD  = 128;   // dim
static constexpr int kStr = 132;  // padded LDS row stride: 16B-aligned rows,
                                  // stride mod 64 dwords = 4 -> conflict-free b64 frags
static constexpr int kChunk = 64; // d rows staged per iteration
static constexpr int kNBuf  = USE_ASYNC ? 2 : 1;

#define NEGV  (-9999.0f)
#define EPSV  (1e-12f)

__device__ __forceinline__ void issue_chunk_async(const float* g, float* l) {
#if USE_ASYNC
    const int tid = threadIdx.x;
#pragma unroll
    for (int i = 0; i < 8; ++i) {              // 2048 B128 moves / 256 threads
        int f   = tid + (i << 8);
        int row = f >> 5;                      // 32 float4 per 128-float row
        int c4  = f & 31;
        __builtin_amdgcn_global_load_async_to_lds_b128(
            (gv4i*)(g + row * kD   + c4 * 4),
            (lv4i*)(l + row * kStr + c4 * 4), 0, 0);
    }
#else
    (void)g; (void)l;
#endif
}

// One block per (n, b); loops over tensor t in {0,1}.
// Writes scores_ws[t*B*N + b*N + n].
__global__ __launch_bounds__(256)
void maxsim_kernel(const float* __restrict__ q_reps,
                   const float* __restrict__ d_cq,
                   const float* __restrict__ d_orig,
                   const int*   __restrict__ d_mask,
                   float* __restrict__ scores_ws) {
    __shared__ float Q[kLq * kStr];               // 16.9 KB
    __shared__ float Dt[kNBuf][kChunk * kStr];    // 33.8 KB x bufs
    __shared__ float invq[kLq];
    __shared__ float colscale[kChunk];
    __shared__ int   colmask[kChunk];
    __shared__ float redmax[kLq * 64];            // 8 KB

    const int x = blockIdx.x;
    const int n = x >> 7;          // x / kB (kB == 128)
    const int b = x & (kB - 1);

    const int tid  = threadIdx.x;
    const int lane = tid & 31;
    const int wave = tid >> 5;
    const int qt = wave & 1;       // q tile 0/1
    const int dt = wave >> 1;      // d tile 0..3 within chunk
    const int hi = lane >> 4;      // K-half select for WMMA fragments
    const int lo = lane & 15;      // row/col within tile

    // ---- stage Q (32x128 f32) once ----
    {
        const float4* qg = (const float4*)(q_reps + (size_t)b * kLq * kD);
#pragma unroll
        for (int i = 0; i < 4; ++i) {
            int f   = tid + (i << 8);
            int row = f >> 5;
            int c4  = f & 31;
            *(float4*)&Q[row * kStr + c4 * 4] = qg[f];
        }
    }
    __syncthreads();
    if (tid < kLq) {                              // per-q-row inverse norms
        float ss = 0.f;
        const float* r = &Q[tid * kStr];
        for (int k = 0; k < kD; ++k) ss += r[k] * r[k];
        invq[tid] = 1.0f / fmaxf(sqrtf(ss), EPSV);
    }
    __syncthreads();

    const size_t dbase = ((size_t)n * kB + b) * (size_t)kLd * kD;
    const int    mbase = (n * kB + b) * kLd;

    for (int t = 0; t < 2; ++t) {
        const float* dten = (t == 0) ? d_cq : d_orig;
        const float* dch  = dten + dbase;         // chunk c at dch + c*kChunk*kD

        float rowmax[8];
#pragma unroll
        for (int i = 0; i < 8; ++i) rowmax[i] = -3.0e38f;

#if USE_ASYNC
        issue_chunk_async(dch, Dt[0]);            // prime buffer 0
#endif
        for (int c = 0; c < 4; ++c) {
            const int buf = USE_ASYNC ? (c & 1) : 0;
#if USE_ASYNC
            if (c < 3) {                          // overlap next chunk's HBM fetch
                issue_chunk_async(dch + (size_t)(c + 1) * kChunk * kD,
                                  Dt[(c + 1) & 1]);
                __builtin_amdgcn_s_wait_asynccnt(8);   // chunk c landed (this wave)
            } else {
                __builtin_amdgcn_s_wait_asynccnt(0);
            }
            __syncthreads();                      // all waves' chunk c visible
            {   // ---- row norms from LDS ----
                int r = tid >> 2, sub = tid & 3;
                const float* lrow = &Dt[buf][r * kStr];
                float ss = 0.f;
#pragma unroll
                for (int i = 0; i < 8; ++i) {
                    float4 v = *(const float4*)&lrow[(sub + 4 * i) * 4];
                    ss += v.x*v.x + v.y*v.y + v.z*v.z + v.w*v.w;
                }
                ss += __shfl_xor(ss, 1, 4);
                ss += __shfl_xor(ss, 2, 4);
                if (sub == 0) {
                    int m = d_mask[mbase + c * kChunk + r];
                    colmask[r]  = m;
                    colscale[r] = (m != 0) ? (1.0f / fmaxf(sqrtf(ss), EPSV)) : 0.0f;
                }
            }
#else
            {   // ---- synchronous stage + fused row sum-of-squares ----
                int r = tid >> 2, sub = tid & 3;
                const float4* dg =
                    (const float4*)(dch + (size_t)(c * kChunk + r) * kD);
                float ss = 0.f;
#pragma unroll
                for (int i = 0; i < 8; ++i) {
                    int c4 = sub + i * 4;
                    float4 v = dg[c4];
                    ss += v.x*v.x + v.y*v.y + v.z*v.z + v.w*v.w;
                    *(float4*)&Dt[0][r * kStr + c4 * 4] = v;
                }
                ss += __shfl_xor(ss, 1, 4);
                ss += __shfl_xor(ss, 2, 4);
                if (sub == 0) {
                    int m = d_mask[mbase + c * kChunk + r];
                    colmask[r]  = m;
                    colscale[r] = (m != 0) ? (1.0f / fmaxf(sqrtf(ss), EPSV)) : 0.0f;
                }
            }
#endif
            __syncthreads();

            // ---- WMMA: tile (qt x dt), K = 128, fp32 ----
            const float* Arow = &Q[(qt * 16 + lo) * kStr];
            const float* Brow = &Dt[buf][(dt * 16 + lo) * kStr];
            v8f acc = {};
#if __has_builtin(__builtin_amdgcn_wmma_f32_16x16x4_f32)
            for (int k0 = 0; k0 < kD; k0 += 4) {
                int kk = k0 + 2 * hi;   // lanes 0-15: K={k0,k0+1}; 16-31: {k0+2,k0+3}
                v2f a  = *(const v2f*)&Arow[kk];
                v2f bb = *(const v2f*)&Brow[kk];
                acc = __builtin_amdgcn_wmma_f32_16x16x4_f32(
                    false, a, false, bb, (short)0, acc, false, false);
            }
#else
            for (int k0 = 0; k0 < kD; k0 += 32) {
                v16bf a, bb;
#pragma unroll
                for (int j = 0; j < 16; ++j) {
                    int k = k0 + j + 8 * hi + ((j >= 8) ? 8 : 0);
                    a[j]  = (__bf16)Arow[k];
                    bb[j] = (__bf16)Brow[k];
                }
                acc = __builtin_amdgcn_wmma_f32_16x16x32_bf16(
                    false, a, false, bb, (short)0, acc, false, false);
            }
#endif
            // ---- fold in 1/||q||, 1/||d||, mask; running row max ----
            int   col = dt * 16 + lo;
            float cs  = colscale[col];
            int   cm  = colmask[col];
#pragma unroll
            for (int i = 0; i < 8; ++i) {
                int qrow = qt * 16 + hi * 8 + i;
                float val = cm ? acc[i] * invq[qrow] * cs : NEGV;
                rowmax[i] = fmaxf(rowmax[i], val);
            }
            __syncthreads();   // protect Dt/colscale before next stage/issue
        }

        // ---- cross-wave max over Ld, then sum over Lq ----
#pragma unroll
        for (int i = 0; i < 8; ++i) {
            int qrow = qt * 16 + hi * 8 + i;
            redmax[qrow * 64 + dt * 16 + lo] = rowmax[i];
        }
        __syncthreads();
        if (wave == 0) {
            float m = -3.0e38f;
            for (int j = 0; j < 64; ++j) m = fmaxf(m, redmax[lane * 64 + j]);
            float s = m;
            for (int off = 16; off > 0; off >>= 1) s += __shfl_xor(s, off, 32);
            if (lane == 0)
                scores_ws[(size_t)t * kB * kN + (size_t)b * kN + n] = s;
        }
        __syncthreads();       // redmax reused by next t
    }
}

// Single block: per-row (B=128) log_softmax over N=8 for student & teacher,
// KL(log_target=true), batchmean.
__global__ __launch_bounds__(128)
void kl_kernel(const float* __restrict__ scores_ws, float* __restrict__ out) {
    __shared__ float red[128];
    int b = threadIdx.x;
    const float* s  = scores_ws + (size_t)b * kN;
    const float* tt = scores_ws + (size_t)kB * kN + (size_t)b * kN;
    float sv[kN], tv[kN];
    float ms = -3.0e38f, mt = -3.0e38f;
    for (int i = 0; i < kN; ++i) {
        sv[i] = s[i]; tv[i] = tt[i];
        ms = fmaxf(ms, sv[i]); mt = fmaxf(mt, tv[i]);
    }
    float es = 0.f, et = 0.f;
    for (int i = 0; i < kN; ++i) { es += expf(sv[i] - ms); et += expf(tv[i] - mt); }
    float lses = logf(es), lset = logf(et);
    float kl = 0.f;
    for (int i = 0; i < kN; ++i) {
        float ls = sv[i] - ms - lses;
        float lt = tv[i] - mt - lset;
        kl += expf(lt) * (lt - ls);
    }
    red[b] = kl;
    __syncthreads();
    for (int off = 64; off > 0; off >>= 1) {
        if (b < off) red[b] += red[b + off];
        __syncthreads();
    }
    if (b == 0) out[0] = red[0] / (float)kB;
}

extern "C" void kernel_launch(void* const* d_in, const int* in_sizes, int n_in,
                              void* d_out, int out_size, void* d_ws, size_t ws_size,
                              hipStream_t stream) {
    (void)in_sizes; (void)n_in; (void)out_size; (void)ws_size;
    const float* q     = (const float*)d_in[0];
    const float* dcq   = (const float*)d_in[1];
    const float* dorig = (const float*)d_in[2];
    const int*   mask  = (const int*)d_in[3];
    // d_in[4] (labels) is unused by the reference loss.
    float* sw = (float*)d_ws;                 // 2 * B * N floats of scratch

    maxsim_kernel<<<dim3(kN * kB), 256, 0, stream>>>(q, dcq, dorig, mask, sw);
    kl_kernel<<<1, 128, 0, stream>>>(sw, (float*)d_out);
}
